// GATConv_62354335203669
// MI455X (gfx1250) — matrix operand order repaired and checked by reading, not compiled, not run on hardware
//
#include <hip/hip_runtime.h>
#include <hip/hip_bf16.h>
#include <math.h>

typedef __attribute__((ext_vector_type(16))) _Float16 v16h;
typedef __attribute__((ext_vector_type(8)))  _Float16 v8h;
typedef __attribute__((ext_vector_type(8)))  float    v8f;

#define HC 256        // H*C
#define INCH 256      // in_channels
#define NHEAD 8
#define NEG_SLOPE 0.2f

// ---------------------------------------------------------------------------
// utility: fill a float buffer
// ---------------------------------------------------------------------------
__global__ void fill_f32(float* __restrict__ p, float v, long long n) {
    long long i = (long long)blockIdx.x * blockDim.x + threadIdx.x;
    long long stride = (long long)gridDim.x * blockDim.x;
    for (; i < n; i += stride) p[i] = v;
}

// ---------------------------------------------------------------------------
// GEMM: X0[N,256] = x[N,256] @ W[256,256]   (f32 in, f16 WMMA, f32 out)
// block = 256 threads (8 waves). Block computes 64 rows x 256 cols.
// Wave w: row-tile rt = w>>1, col-tiles (w&1)*8 .. +7. K split into 8 steps
// of 32 with a double-buffered W^T slab (one barrier per K-step).
// ---------------------------------------------------------------------------
__global__ __launch_bounds__(256) void gemm_x_w(const float* __restrict__ x,
                                                const float* __restrict__ W,
                                                float* __restrict__ X0,
                                                int Nrows) {
    __shared__ _Float16 Als[64][264];      // A rows (f16), 528B stride (16B mult)
    __shared__ _Float16 Bls[2][256][40];   // W^T slabs: [buf][n][k], 80B stride

    const int t      = threadIdx.x;
    const int wave   = t >> 5;
    const int lane   = t & 31;
    const int lanelo = lane & 15;
    const int hi     = lane >> 4;          // 0 or 1 (upper half-wave)
    const int block_row = blockIdx.x * 64;

    // ---- stage A block: 64 rows x 256 cols, f32 -> f16 (zero-pad OOB rows)
    for (int i = t; i < 64 * INCH; i += 256) {
        int r = i >> 8, c = i & 255;
        int gr = block_row + r;
        float v = (gr < Nrows) ? x[(long long)gr * INCH + c] : 0.0f;
        Als[r][c] = (_Float16)v;
    }

    // ---- stage W^T slab 0: W[kl][n] -> Bls[0][n][kl]
    for (int i = t; i < 32 * HC; i += 256) {
        int kl = i >> 8, n = i & 255;       // consecutive t -> consecutive n
        Bls[0][n][kl] = (_Float16)W[(long long)kl * HC + n];
    }
    __syncthreads();

    const int rt     = wave >> 1;
    const int ctbase = (wave & 1) * 8;
    const int arow   = rt * 16 + lanelo;

    v8f acc[8];
#pragma unroll
    for (int j = 0; j < 8; ++j) acc[j] = (v8f){};

    for (int ks = 0; ks < 8; ++ks) {
        const int cur = ks & 1;
        // prefetch-stage next slab into the other buffer (overlaps WMMA work)
        if (ks < 7) {
            const int nxt = cur ^ 1;
            for (int i = t; i < 32 * HC; i += 256) {
                int kl = i >> 8, n = i & 255;
                Bls[nxt][n][kl] = (_Float16)W[(long long)((ks + 1) * 32 + kl) * HC + n];
            }
        }

        // A fragment: elems 0-7 = K (hi?8:0)+0..7 ; elems 8-15 = +16
        const int k0 = ks * 32;
        const v8h* a0 = (const v8h*)&Als[arow][k0 + hi * 8];
        const v8h* a1 = (const v8h*)&Als[arow][k0 + 16 + hi * 8];
        v16h a = __builtin_shufflevector(a0[0], a1[0],
                 0,1,2,3,4,5,6,7,8,9,10,11,12,13,14,15);

        // load all 8 B fragments first so the ds_loads clause & overlap,
        // then issue the 8 WMMAs back-to-back
        v16h bfr[8];
#pragma unroll
        for (int j = 0; j < 8; ++j) {
            const int n = (ctbase + j) * 16 + lanelo;   // lane col
            const v8h* b0 = (const v8h*)&Bls[cur][n][hi * 8];
            const v8h* b1 = (const v8h*)&Bls[cur][n][16 + hi * 8];
            bfr[j] = __builtin_shufflevector(b0[0], b1[0],
                     0,1,2,3,4,5,6,7,8,9,10,11,12,13,14,15);
        }
#pragma unroll
        for (int j = 0; j < 8; ++j) {
            acc[j] = __builtin_amdgcn_wmma_f32_16x16x32_f16(
                         false, a, false, bfr[j], (short)0, acc[j], false, false);
        }
        __syncthreads();   // staging of ks+1 done; readers of cur done
    }

    // ---- store: VGPR v -> row (v + 8*hi), col lanelo
    const int srow = block_row + rt * 16 + hi * 8;
#pragma unroll
    for (int j = 0; j < 8; ++j) {
        const int n = (ctbase + j) * 16 + lanelo;
#pragma unroll
        for (int v = 0; v < 8; ++v) {
            int gr = srow + v;
            if (gr < Nrows) X0[(long long)gr * HC + n] = acc[j][v];
        }
    }
}

// ---------------------------------------------------------------------------
// scatter-mean numerators: Xe_sum[e][c] += X0[v][c]; cnt[e] += 1
// one block per incidence, thread = channel
// ---------------------------------------------------------------------------
__global__ __launch_bounds__(256) void scatter_edge_sum(const float* __restrict__ X0,
                                                        const int* __restrict__ vertex,
                                                        const int* __restrict__ edges,
                                                        float* __restrict__ Xe_sum,
                                                        float* __restrict__ cnt,
                                                        int nnz) {
    int i = blockIdx.x;
    if (i >= nnz) return;
    int v = vertex[i];
    int e = edges[i];
    float val = X0[(long long)v * HC + threadIdx.x];
    atomicAdd(&Xe_sum[(long long)e * HC + threadIdx.x], val);
    if (threadIdx.x == 0) atomicAdd(&cnt[e], 1.0f);
}

// ---------------------------------------------------------------------------
// per-edge: mean, attention logit per head, head-dim L2 norm output
// block per edge, 256 threads, thread t -> (h = t>>5, c = t&31)
// ---------------------------------------------------------------------------
__global__ __launch_bounds__(256) void edge_finalize(float* __restrict__ Xe,  // in: sums, out: means (in place)
                                                     const float* __restrict__ cnt,
                                                     const float* __restrict__ att_e,
                                                     float* __restrict__ alpha_e,
                                                     float* __restrict__ Xe_n) {
    __shared__ float xs[HC];
    const int e = blockIdx.x;
    const int t = threadIdx.x;
    float inv = 1.0f / fmaxf(cnt[e], 1.0f);
    float val = Xe[(long long)e * HC + t] * inv;
    Xe[(long long)e * HC + t] = val;
    xs[t] = val;

    // per-head dot with att_e: one wave == one head (t>>5 uniform per wave)
    float p = val * att_e[t];
#pragma unroll
    for (int off = 16; off > 0; off >>= 1) p += __shfl_down(p, off, 32);
    if ((t & 31) == 0) alpha_e[e * NHEAD + (t >> 5)] = p;
    __syncthreads();

    // L2 norm over heads for fixed c (reference: axis=1 of [E,H,C])
    float ss = 0.0f;
#pragma unroll
    for (int h = 0; h < NHEAD; ++h) {
        float z = xs[h * 32 + (t & 31)];
        ss += z * z;
    }
    float nrm = sqrtf(ss);
    float scale = (nrm > 0.0f) ? (1.0f / nrm) : 0.0f;
    Xe_n[(long long)e * HC + t] = val * scale;
}

// ---------------------------------------------------------------------------
// float atomic max via int punning (works with -inf init)
// ---------------------------------------------------------------------------
__device__ inline void atomicMaxFloat(float* addr, float val) {
    int bits = __float_as_int(val);
    if (bits >= 0)
        atomicMax((int*)addr, bits);
    else
        atomicMin((unsigned int*)addr, (unsigned int)bits);
}

__device__ inline float leaky(float a) { return a >= 0.0f ? a : NEG_SLOPE * a; }

// ---------------------------------------------------------------------------
// segment max of leaky(logit) over incidences grouped by vertex
// ---------------------------------------------------------------------------
__global__ void seg_logit_max(const float* __restrict__ alpha_e,
                              const int* __restrict__ vertex,
                              const int* __restrict__ edges,
                              float* __restrict__ m, int nnz) {
    long long total = (long long)nnz * NHEAD;
    long long stride = (long long)gridDim.x * blockDim.x;
    for (long long tid = (long long)blockIdx.x * blockDim.x + threadIdx.x;
         tid < total; tid += stride) {
        int i = (int)(tid >> 3);
        int h = (int)(tid & 7);
        float a = leaky(alpha_e[edges[i] * NHEAD + h]);
        atomicMaxFloat(&m[vertex[i] * NHEAD + h], a);
    }
}

// segment sum of exp(a - m)
__global__ void seg_expsum(const float* __restrict__ alpha_e,
                           const int* __restrict__ vertex,
                           const int* __restrict__ edges,
                           const float* __restrict__ m,
                           float* __restrict__ s, int nnz) {
    long long total = (long long)nnz * NHEAD;
    long long stride = (long long)gridDim.x * blockDim.x;
    for (long long tid = (long long)blockIdx.x * blockDim.x + threadIdx.x;
         tid < total; tid += stride) {
        int i = (int)(tid >> 3);
        int h = (int)(tid & 7);
        int v = vertex[i];
        float a = leaky(alpha_e[edges[i] * NHEAD + h]);
        atomicAdd(&s[v * NHEAD + h], expf(a - m[v * NHEAD + h]));
    }
}

// ---------------------------------------------------------------------------
// weighted scatter back to nodes: Xv[v][c] += Xe[e][c] * alpha(i,h)
// block per incidence; alpha recomputed (cheaper than a 16MB ex buffer)
// ---------------------------------------------------------------------------
__global__ __launch_bounds__(256) void scatter_node_sum(const float* __restrict__ Xe,
                                                        const float* __restrict__ alpha_e,
                                                        const int* __restrict__ vertex,
                                                        const int* __restrict__ edges,
                                                        const float* __restrict__ m,
                                                        const float* __restrict__ s,
                                                        float* __restrict__ Xv,
                                                        int nnz) {
    int i = blockIdx.x;
    if (i >= nnz) return;
    int v = vertex[i];
    int e = edges[i];
    int t = threadIdx.x;
    int h = t >> 5;
    float a  = leaky(alpha_e[e * NHEAD + h]);
    float al = expf(a - m[v * NHEAD + h]) / (s[v * NHEAD + h] + 1e-16f);
    atomicAdd(&Xv[(long long)v * HC + t], Xe[(long long)e * HC + t] * al);
}

// ---------------------------------------------------------------------------
// in-place row L2 normalize of Xv[N,256]
// ---------------------------------------------------------------------------
__global__ __launch_bounds__(256) void node_normalize(float* __restrict__ Xv, int Nrows) {
    __shared__ float red[8];
    __shared__ float scale_s;
    int n = blockIdx.x;
    if (n >= Nrows) return;
    int t = threadIdx.x;
    float val = Xv[(long long)n * HC + t];
    float ss = val * val;
#pragma unroll
    for (int off = 16; off > 0; off >>= 1) ss += __shfl_down(ss, off, 32);
    if ((t & 31) == 0) red[t >> 5] = ss;
    __syncthreads();
    if (t == 0) {
        float tot = 0.0f;
#pragma unroll
        for (int w = 0; w < 8; ++w) tot += red[w];
        float nrm = sqrtf(tot);
        scale_s = (nrm > 0.0f) ? (1.0f / nrm) : 0.0f;
    }
    __syncthreads();
    Xv[(long long)n * HC + t] = val * scale_s;
}

// ---------------------------------------------------------------------------
extern "C" void kernel_launch(void* const* d_in, const int* in_sizes, int n_in,
                              void* d_out, int out_size, void* d_ws, size_t ws_size,
                              hipStream_t stream) {
    const float* x     = (const float*)d_in[0];   // [N,256]
    const float* W     = (const float*)d_in[1];   // [256,256]
    const float* att_e = (const float*)d_in[2];   // [256]
    const int*   hidx  = (const int*)d_in[3];     // [2,NNZ]

    const int N   = in_sizes[0] / INCH;
    const int NNZ = in_sizes[3] / 2;
    const int E   = out_size / HC - N;

    const int* vertex = hidx;
    const int* edges  = hidx + NNZ;

    // workspace layout (floats)
    float* X0      = (float*)d_ws;                 // N*256
    float* Xe      = X0 + (size_t)N * HC;          // E*256 (sums -> means)
    float* cnt     = Xe + (size_t)E * HC;          // E
    float* alpha_e = cnt + E;                      // E*8
    float* m       = alpha_e + (size_t)E * NHEAD;  // N*8
    float* s       = m + (size_t)N * NHEAD;        // N*8

    float* Xv   = (float*)d_out;                   // [N,256] (accum in place)
    float* Xe_n = Xv + (size_t)N * HC;             // [E,256]

    (void)n_in; (void)ws_size;

    // init accumulators
    fill_f32<<<2048, 256, 0, stream>>>(Xv, 0.0f, (long long)N * HC);
    fill_f32<<<1024, 256, 0, stream>>>(Xe, 0.0f, (long long)E * HC);
    fill_f32<<<64,   256, 0, stream>>>(cnt, 0.0f, (long long)E);
    fill_f32<<<256,  256, 0, stream>>>(m, -INFINITY, (long long)N * NHEAD);
    fill_f32<<<256,  256, 0, stream>>>(s, 0.0f, (long long)N * NHEAD);

    // dense projection on the matrix engine
    gemm_x_w<<<(N + 63) / 64, 256, 0, stream>>>(x, W, X0, N);

    // scatter-mean into hyperedges
    scatter_edge_sum<<<NNZ, 256, 0, stream>>>(X0, vertex, edges, Xe, cnt, NNZ);
    edge_finalize<<<E, 256, 0, stream>>>(Xe, cnt, att_e, alpha_e, Xe_n);

    // segment softmax over vertex groups
    seg_logit_max<<<4096, 256, 0, stream>>>(alpha_e, vertex, edges, m, NNZ);
    seg_expsum  <<<4096, 256, 0, stream>>>(alpha_e, vertex, edges, m, s, NNZ);

    // weighted scatter back to nodes + row normalize
    scatter_node_sum<<<NNZ, 256, 0, stream>>>(Xe, alpha_e, vertex, edges, m, s, Xv, NNZ);
    node_normalize<<<N, 256, 0, stream>>>(Xv, N);
}
